// _SparseAttention_1571958030955
// MI455X (gfx1250) — matrix operand order repaired and checked by reading, not compiled, not run on hardware
//
#include <hip/hip_runtime.h>
#include <hip/hip_bf16.h>

// ---------------------------------------------------------------------------
// Top-k (k=32) sparse attention for MI455X / gfx1250, bf16 WMMA pipeline.
//   B=2, T=2048, D=1024, H=16, DH=64, TOPK=32
// Dense matmuls (QKV proj, QK^T scores, output proj) -> v_wmma_f32_16x16x32_bf16
// with fragments built from contiguous 16B vector loads (b128), per the CDNA5
// A/B lane layouts. GEMM staging uses GLOBAL_LOAD_ASYNC_TO_LDS_B128 (ASYNCcnt)
// with double buffering so DMA overlaps WMMA compute. AV uses true top-32
// sparsity (gather). Whole problem (~80MB) fits in the 192MB L2.
// ---------------------------------------------------------------------------

#define B_    2
#define T_    2048
#define D_    1024
#define H_    16
#define DH_   64
#define TOPK_ 32

typedef __attribute__((ext_vector_type(16))) __bf16 v16bf;
typedef __attribute__((ext_vector_type(8)))  __bf16 v8bf;
typedef __attribute__((ext_vector_type(8)))  float  v8f;

__device__ __forceinline__ unsigned short f2bf(float f) {
    union { float f; unsigned u; } x; x.f = f;
    unsigned r = x.u + 0x7FFFu + ((x.u >> 16) & 1u);   // round-to-nearest-even
    return (unsigned short)(r >> 16);
}
__device__ __forceinline__ float bf2f(unsigned short h) {
    union { unsigned u; float f; } y; y.u = ((unsigned)h) << 16;
    return y.f;
}

// A fragment: lane group g reads K runs [ks+8g, +8) and [ks+16+8g, +8)
// from a row-major (K-contiguous) row pointer. Two 16B vector loads.
__device__ __forceinline__ v16bf frag_a(const unsigned short* rowp, int ks, int g) {
    v8bf lo = *(const v8bf*)(rowp + ks + 8 * g);
    v8bf hi = *(const v8bf*)(rowp + ks + 16 + 8 * g);
    return __builtin_shufflevector(lo, hi, 0,1,2,3,4,5,6,7,8,9,10,11,12,13,14,15);
}
// B fragment (B = W^T): lane group g reads K runs [ks+8g, +8) and [ks+16+8g, +8) from W's row.
__device__ __forceinline__ v16bf frag_b(const unsigned short* rowp, int ks, int g) {
    v8bf lo = *(const v8bf*)(rowp + ks + 8 * g);
    v8bf hi = *(const v8bf*)(rowp + ks + 16 + 8 * g);
    return __builtin_shufflevector(lo, hi, 0,1,2,3,4,5,6,7,8,9,10,11,12,13,14,15);
}

// CDNA5 async global->LDS DMA (no VGPR round-trip), tracked by ASYNCcnt.
// lds = LDS byte offset (low 32 bits of a generic pointer into __shared__).
__device__ __forceinline__ void async_copy_b128(unsigned lds, const void* gptr) {
    asm volatile("global_load_async_to_lds_b128 %0, %1, off"
                 :: "v"(lds), "v"(gptr) : "memory");
}
__device__ __forceinline__ void wait_asynccnt0() {
    asm volatile("s_wait_asynccnt 0x0" ::: "memory");
}

// ---------------------------------------------------------------------------
// f32 -> bf16 conversion, vectorized: 8 elems/thread (2x float4 -> 1x uint4).
// n must be a multiple of 8 (all our sizes are multiples of 2048).
// ---------------------------------------------------------------------------
__global__ void cvt_f32_bf16(const float* __restrict__ in,
                             unsigned short* __restrict__ out, size_t n) {
    size_t i = ((size_t)blockIdx.x * blockDim.x + threadIdx.x) * 8;
    if (i >= n) return;
    float4 f0 = *(const float4*)(in + i);
    float4 f1 = *(const float4*)(in + i + 4);
    uint4 u;
    u.x = ((unsigned)f2bf(f0.y) << 16) | f2bf(f0.x);
    u.y = ((unsigned)f2bf(f0.w) << 16) | f2bf(f0.z);
    u.z = ((unsigned)f2bf(f1.y) << 16) | f2bf(f1.x);
    u.w = ((unsigned)f2bf(f1.w) << 16) | f2bf(f1.z);
    *(uint4*)(out + i) = u;
}

// ---------------------------------------------------------------------------
// O = A(MxK) @ W(NxK)^T * scale + bias, bf16 in, f32 accumulate.
// Block tile 128x64, 256 threads = 8 waves laid 4(M) x 2(N); each wave owns
// 32x32 -> 4 accumulators, 4 WMMAs per 32-wide k-step. Double-buffered LDS
// tiles filled by async global->LDS DMA, overlapping the WMMA compute.
// ---------------------------------------------------------------------------
#define LPITCH 72   // LDS row pitch in halfs (144B, multiple of 16B)

__global__ __launch_bounds__(256) void gemm_bf16_wmma(
    const unsigned short* __restrict__ A,   // M x K bf16 row-major
    const unsigned short* __restrict__ W,   // N x K bf16 row-major
    const float* __restrict__ bias,         // N
    unsigned short* __restrict__ out_bf,    // optional bf16 output
    float* __restrict__ out_f32,            // optional f32 output
    int M, int N, int K, float scale, int bhtd)
{
    __shared__ unsigned short As[2][128][LPITCH];   // 2 x 18KB
    __shared__ unsigned short Ws[2][64][LPITCH];    // 2 x  9KB

    const int tid  = threadIdx.x;
    const int lane = tid & 31;
    const int wid  = tid >> 5;
    const int g    = lane >> 4;
    const int ln   = lane & 15;
    const int m0   = blockIdx.y * 128;
    const int n0   = blockIdx.x * 64;
    const int wm   = (wid & 3) * 32;    // wave M offset in tile
    const int wn   = (wid >> 2) * 32;   // wave N offset in tile

    // per-thread staging assignment (whole block covers both tiles)
    const int arow = tid >> 1, acol = (tid & 1) * 32;   // A: 128 rows x 64 halfs
    const int wrow = tid >> 2, wcol = (tid & 3) * 16;   // W:  64 rows x 64 halfs

    // issue async DMA of tile at column kb into buffer `buf`
    auto stage = [&](int buf, int kb) {
        const unsigned short* asrc = A + (size_t)(m0 + arow) * K + kb + acol;
        unsigned al = (unsigned)(size_t)&As[buf][arow][acol];
        async_copy_b128(al,      asrc);
        async_copy_b128(al + 16, asrc + 8);
        async_copy_b128(al + 32, asrc + 16);
        async_copy_b128(al + 48, asrc + 24);
        const unsigned short* wsrc = W + (size_t)(n0 + wrow) * K + kb + wcol;
        unsigned wl = (unsigned)(size_t)&Ws[buf][wrow][wcol];
        async_copy_b128(wl,      wsrc);
        async_copy_b128(wl + 16, wsrc + 8);
    };

    v8f acc[2][2] = {};
    int buf = 0;
    stage(0, 0);

    for (int kb = 0; kb < K; kb += 64) {
        wait_asynccnt0();       // my DMA into `buf` has landed in LDS
        __syncthreads();        // everyone's has; previous compute also done
        if (kb + 64 < K) stage(buf ^ 1, kb + 64);   // prefetch next tile

#pragma unroll
        for (int ks = 0; ks < 64; ks += 32) {
            v16bf a0 = frag_a(&As[buf][wm + ln][0],      ks, g);
            v16bf a1 = frag_a(&As[buf][wm + 16 + ln][0], ks, g);
            v16bf b0 = frag_b(&Ws[buf][wn + ln][0],      ks, g);
            v16bf b1 = frag_b(&Ws[buf][wn + 16 + ln][0], ks, g);
            acc[0][0] = __builtin_amdgcn_wmma_f32_16x16x32_bf16(false, a0, false, b0, (short)0, acc[0][0], false, false);
            acc[0][1] = __builtin_amdgcn_wmma_f32_16x16x32_bf16(false, a0, false, b1, (short)0, acc[0][1], false, false);
            acc[1][0] = __builtin_amdgcn_wmma_f32_16x16x32_bf16(false, a1, false, b0, (short)0, acc[1][0], false, false);
            acc[1][1] = __builtin_amdgcn_wmma_f32_16x16x32_bf16(false, a1, false, b1, (short)0, acc[1][1], false, false);
        }
        buf ^= 1;
    }

    // epilogue: C layout -> lane ln is column, vgpr i is row i + 8*g
#pragma unroll
    for (int mi = 0; mi < 2; ++mi) {
#pragma unroll
        for (int ni = 0; ni < 2; ++ni) {
#pragma unroll
            for (int i = 0; i < 8; ++i) {
                int m = m0 + wm + mi * 16 + i + 8 * g;
                int n = n0 + wn + ni * 16 + ln;
                float v = acc[mi][ni][i] * scale + bias[n];
                if (out_f32) out_f32[(size_t)m * N + n] = v;
                if (out_bf) {
                    size_t idx;
                    if (bhtd) {   // (m = b*T+t, n = h*DH+dh) -> (B,H,T,DH)
                        int b  = m >> 11;          // / T_
                        int t  = m & (T_ - 1);
                        int hh = n >> 6;           // / DH_
                        int dh = n & (DH_ - 1);
                        idx = ((size_t)((b * H_ + hh) * T_ + t)) * DH_ + dh;
                    } else {
                        idx = (size_t)m * N + n;
                    }
                    out_bf[idx] = f2bf(v);
                }
            }
        }
    }
}

// ---------------------------------------------------------------------------
// Per-(b,h) attention over a 16-query tile. 512 threads = 16 waves.
//   Phase 1: scores 16x2048 via WMMA (fragments = b128 global loads),
//            stored f32 in a 128KB LDS buffer (320KB/WGP on CDNA5).
//   Phase 2: wave w owns row w: 32 argmax rounds -> exact top-32.
//   Phase 3: softmax over 32 winners, sparse-gather AV (lane owns 2 dims).
// Q is pre-scaled by 1/sqrt(DH) at projection time.
// ---------------------------------------------------------------------------
__global__ __launch_bounds__(512) void topk_attn(
    const unsigned short* __restrict__ Q,    // (B,H,T,DH) bf16, pre-scaled
    const unsigned short* __restrict__ Kb,   // (B,H,T,DH) bf16
    const unsigned short* __restrict__ Vb,   // (B,H,T,DH) bf16
    unsigned short* __restrict__ ctx)        // (B,T,D) bf16
{
    __shared__ float sc[16][T_];             // 131072 B
    __shared__ float topv[16][TOPK_];        //   2048 B
    __shared__ int   topi[16][TOPK_];        //   2048 B

    const int lane = threadIdx.x & 31;
    const int w    = threadIdx.x >> 5;       // wave id = owned query row
    const int g    = lane >> 4;
    const int ln   = lane & 15;
    const int q0   = blockIdx.x * 16;
    const int h    = blockIdx.y;
    const int b    = blockIdx.z;
    const size_t headbase = ((size_t)(b * H_ + h)) * T_ * DH_;

    // ---- Phase 1: scores via WMMA -------------------------------------
    const unsigned short* qrow = Q + headbase + (size_t)(q0 + ln) * DH_;
    v16bf aq0 = frag_a(qrow, 0, g);
    v16bf aq1 = frag_a(qrow, 32, g);

    for (int kt = 0; kt < T_; kt += 256) {       // 16 waves x 16 keys = 256/iter
        int key0 = kt + w * 16;
        const unsigned short* krow = Kb + headbase + (size_t)(key0 + ln) * DH_;
        v8f s = {};
        v16bf bk0 = frag_b(krow, 0, g);
        v16bf bk1 = frag_b(krow, 32, g);
        s = __builtin_amdgcn_wmma_f32_16x16x32_bf16(false, aq0, false, bk0, (short)0, s, false, false);
        s = __builtin_amdgcn_wmma_f32_16x16x32_bf16(false, aq1, false, bk1, (short)0, s, false, false);
#pragma unroll
        for (int i = 0; i < 8; ++i) sc[i + 8 * g][key0 + ln] = s[i];
    }
    __syncthreads();

    // ---- Phase 2: exact top-32 for row w (one wave per row) -----------
    float* row = sc[w];
    for (int r = 0; r < TOPK_; ++r) {
        float best = -3.0e38f; int bidx = 0;
        for (int j = lane; j < T_; j += 32) {
            float v = row[j];
            if (v > best) { best = v; bidx = j; }
        }
#pragma unroll
        for (int off = 16; off > 0; off >>= 1) {
            float ov = __shfl_xor(best, off, 32);
            int   oi = __shfl_xor(bidx, off, 32);
            if (ov > best || (ov == best && oi < bidx)) { best = ov; bidx = oi; }
        }
        if (lane == 0) {
            topv[w][r] = best;
            topi[w][r] = bidx;
            row[bidx]  = -3.0e38f;   // knock out; same-wave LDS ordering holds
        }
    }

    // ---- Phase 3: softmax over 32 winners, sparse AV gather -----------
    {
        float m = topv[w][0];                     // round 0 is the row max
        float p = __expf(topv[w][lane] - m);
        float sum = p;
#pragma unroll
        for (int off = 16; off > 0; off >>= 1) sum += __shfl_xor(sum, off, 32);
        topv[w][lane] = p / sum;                  // store probabilities
    }

    float a0 = 0.f, a1 = 0.f;                     // lane owns dims (lane, lane+32)
    for (int r = 0; r < TOPK_; ++r) {
        float p  = topv[w][r];
        int  idx = topi[w][r];
        const unsigned short* vr = Vb + headbase + (size_t)idx * DH_;
        a0 += p * bf2f(vr[lane]);
        a1 += p * bf2f(vr[lane + 32]);
    }
    size_t obase = ((size_t)(b * T_ + q0 + w)) * D_ + h * DH_;
    ctx[obase + lane]      = f2bf(a0);
    ctx[obase + lane + 32] = f2bf(a1);
}

// ---------------------------------------------------------------------------
// Host-side orchestration
// ---------------------------------------------------------------------------
extern "C" void kernel_launch(void* const* d_in, const int* in_sizes, int n_in,
                              void* d_out, int out_size, void* d_ws, size_t ws_size,
                              hipStream_t stream) {
    (void)in_sizes; (void)n_in; (void)out_size; (void)ws_size;

    const float* x  = (const float*)d_in[0];
    const float* Wq = (const float*)d_in[1];
    const float* bq = (const float*)d_in[2];
    const float* Wk = (const float*)d_in[3];
    const float* bk = (const float*)d_in[4];
    const float* Wv = (const float*)d_in[5];
    const float* bv = (const float*)d_in[6];
    const float* Wo = (const float*)d_in[7];
    const float* bo = (const float*)d_in[8];
    float* out = (float*)d_out;

    const size_t nx = (size_t)B_ * T_ * D_;   // 4,194,304
    const size_t nw = (size_t)D_ * D_;        // 1,048,576

    // workspace carve-up (bf16 elements): total 24M elems = 48 MB
    unsigned short* p   = (unsigned short*)d_ws;
    unsigned short* xb  = p; p += nx;
    unsigned short* wqb = p; p += nw;
    unsigned short* wkb = p; p += nw;
    unsigned short* wvb = p; p += nw;
    unsigned short* wob = p; p += nw;
    unsigned short* qb  = p; p += nx;
    unsigned short* kb2 = p; p += nx;
    unsigned short* vb2 = p; p += nx;
    unsigned short* ctx = p; p += nx;

    // 1) f32 -> bf16 conversions (vectorized)
    {
        const int TPB = 256, EPT = 8;
        cvt_f32_bf16<<<dim3((unsigned)((nx + TPB*EPT - 1) / (TPB*EPT))), TPB, 0, stream>>>(x,  xb,  nx);
        cvt_f32_bf16<<<dim3((unsigned)((nw + TPB*EPT - 1) / (TPB*EPT))), TPB, 0, stream>>>(Wq, wqb, nw);
        cvt_f32_bf16<<<dim3((unsigned)((nw + TPB*EPT - 1) / (TPB*EPT))), TPB, 0, stream>>>(Wk, wkb, nw);
        cvt_f32_bf16<<<dim3((unsigned)((nw + TPB*EPT - 1) / (TPB*EPT))), TPB, 0, stream>>>(Wv, wvb, nw);
        cvt_f32_bf16<<<dim3((unsigned)((nw + TPB*EPT - 1) / (TPB*EPT))), TPB, 0, stream>>>(Wo, wob, nw);
    }

    // 2) Q/K/V projections (WMMA GEMM); 1/sqrt(DH)=0.125 folded into Q
    dim3 gg(D_ / 64, (B_ * T_) / 128);   // 16 x 32 blocks
    gemm_bf16_wmma<<<gg, 256, 0, stream>>>(xb, wqb, bq, qb,  nullptr, B_*T_, D_, D_, 0.125f, 1);
    gemm_bf16_wmma<<<gg, 256, 0, stream>>>(xb, wkb, bk, kb2, nullptr, B_*T_, D_, D_, 1.0f,   1);
    gemm_bf16_wmma<<<gg, 256, 0, stream>>>(xb, wvb, bv, vb2, nullptr, B_*T_, D_, D_, 1.0f,   1);

    // 3) top-k attention (scores via WMMA, sparse AV gather)
    topk_attn<<<dim3(T_ / 16, H_, B_), 512, 0, stream>>>(qb, kb2, vb2, ctx);

    // 4) output projection -> f32 d_out
    gemm_bf16_wmma<<<gg, 256, 0, stream>>>(ctx, wob, bo, nullptr, out, B_*T_, D_, D_, 1.0f, 0);
}
